// MultiHeadAttention_4690104287475
// MI455X (gfx1250) — matrix operand order repaired and checked by reading, not compiled, not run on hardware
//
#include <hip/hip_runtime.h>
#include <stdint.h>

// MI455X / gfx1250, wave32, WMMA f16 -> f32 accumulate, async global->LDS staging.

typedef __attribute__((ext_vector_type(16))) _Float16 v16h;
typedef __attribute__((ext_vector_type(8)))  float    v8f;

#define B_  4
#define S_  2048
#define D_  1024
#define H_  16
#define DH_ 64
#define BS_ (B_ * S_)

// ---------------------------------------------------------------------------
// Async global->LDS (ASYNCcnt-tracked DMA-style copy, CDNA5 ISA)
// lds_off: byte offset within workgroup LDS (generic-pointer low 32 bits).
// ---------------------------------------------------------------------------
__device__ __forceinline__ void async_ld_b128(uint32_t lds_off, const void* gaddr) {
  asm volatile("global_load_async_to_lds_b128 %0, %1, off"
               :
               : "v"(lds_off), "v"(gaddr)
               : "memory");
}
#define WAIT_ASYNC_0() asm volatile("s_wait_asynccnt 0x0" ::: "memory")
#define WAIT_ASYNC_4() asm volatile("s_wait_asynccnt 0x4" ::: "memory")

// ---------------------------------------------------------------------------
// WMMA fragment loaders (CDNA5 ISA §7.12.2, wave32)
// A (16xK=32, 16-bit): lane -> M = lane&15 ; VGPR i packs K = (i/4)*16 + (lane/16)*8 + (i%4)*2 (+1)
// B (K=32x16, 16-bit): lane -> N = lane&15 ; VGPR i packs K = (i/4)*16 + (lane/16)*8 + (i%4)*2 (+1)
// C/D (16x16 f32):     lane -> N = lane&15 ; VGPR r -> M = r + 8*(lane/16)
// ---------------------------------------------------------------------------
__device__ __forceinline__ v16h load_frag_a(const _Float16* tile, int stride, int lane) {
  const _Float16* row = tile + (size_t)(lane & 15) * stride;
  const int koff = (lane >> 4) << 3;
  v16h f;
#pragma unroll
  for (int i = 0; i < 8; ++i) {
    const int k = ((i >> 2) << 4) + koff + ((i & 3) << 1);
    f[2 * i]     = row[k];
    f[2 * i + 1] = row[k + 1];
  }
  return f;
}

__device__ __forceinline__ v16h load_frag_b(const _Float16* tile, int stride, int lane) {
  const _Float16* src = tile + (size_t)(lane & 15) * stride + ((lane >> 4) << 3);
  v16h f;
#pragma unroll
  for (int e = 0; e < 16; ++e) f[e] = src[((e >> 3) << 4) + (e & 7)];
  return f;
}

// ---------------------------------------------------------------------------
// Elementwise converts
// ---------------------------------------------------------------------------
__global__ __launch_bounds__(256) void cvt_f32_f16(const float* __restrict__ in,
                                                   _Float16* __restrict__ out, int n) {
  const int idx = blockIdx.x * 256 + threadIdx.x;
  if (idx < n) out[idx] = (_Float16)in[idx];
}

// Wt[n][k] = W[k][n]  (so B-fragments load contiguously)
__global__ __launch_bounds__(256) void transpose_cvt(const float* __restrict__ W,
                                                     _Float16* __restrict__ Wt) {
  const int idx = blockIdx.x * 256 + threadIdx.x;  // n*D_ + k
  const int n = idx >> 10;
  const int k = idx & (D_ - 1);
  Wt[idx] = (_Float16)W[(size_t)k * D_ + n];
}

// ---------------------------------------------------------------------------
// Tiled WMMA GEMM: C[M,N] = A[M,K] * Bt[N,K]^T ; block tile 128x128, BK=32,
// 8 waves (2x4), each wave 64x32 (4x2 WMMA accumulators).
// Double-buffered LDS filled by global_load_async_to_lds_b128; while tile t is
// consumed by WMMA, tile t+1's async loads are in flight (s_wait_asynccnt 4).
// ---------------------------------------------------------------------------
template <bool OUT_F16>
__global__ __launch_bounds__(256) void gemm_wmma(const _Float16* __restrict__ A,
                                                 const _Float16* __restrict__ Bt,
                                                 void* __restrict__ Cout,
                                                 int M, int N, int K) {
  // stride 40 f16 = 80B rows: 16B-aligned chunks + conflict-free fragment reads
  __shared__ _Float16 As[2][128][40];
  __shared__ _Float16 Bs[2][128][40];

  const int tid  = threadIdx.x;
  const int lane = tid & 31;
  const int wave = tid >> 5;
  const int wm = wave >> 2;  // 0..1
  const int wn = wave & 3;   // 0..3
  const int blockN = blockIdx.x * 128;
  const int blockM = blockIdx.y * 128;

  const uint32_t ldsA0 = (uint32_t)(uintptr_t)&As[0][0][0];
  const uint32_t ldsB0 = (uint32_t)(uintptr_t)&Bs[0][0][0];
  const uint32_t bufSz = 128u * 40u * 2u;

  const int r0 = tid >> 2;         // rows 0..63
  const int r1 = 64 + (tid >> 2);  // rows 64..127
  const int c4 = tid & 3;          // 16B chunk in row

  const size_t rowBytes = (size_t)K * 2;
  const char* gAbase = (const char*)A + (size_t)blockM * rowBytes;
  const char* gBbase = (const char*)Bt + (size_t)blockN * rowBytes;

  auto stage = [&](int k0, int buf) {
    const char* gA = gAbase + (size_t)k0 * 2;
    const char* gB = gBbase + (size_t)k0 * 2;
    const uint32_t la = ldsA0 + (uint32_t)buf * bufSz;
    const uint32_t lb = ldsB0 + (uint32_t)buf * bufSz;
    async_ld_b128(la + (uint32_t)(r0 * 80 + c4 * 16), gA + (size_t)r0 * rowBytes + c4 * 16);
    async_ld_b128(lb + (uint32_t)(r0 * 80 + c4 * 16), gB + (size_t)r0 * rowBytes + c4 * 16);
    async_ld_b128(la + (uint32_t)(r1 * 80 + c4 * 16), gA + (size_t)r1 * rowBytes + c4 * 16);
    async_ld_b128(lb + (uint32_t)(r1 * 80 + c4 * 16), gB + (size_t)r1 * rowBytes + c4 * 16);
  };

  v8f acc[4][2];
#pragma unroll
  for (int i = 0; i < 4; ++i)
#pragma unroll
    for (int j = 0; j < 2; ++j)
#pragma unroll
      for (int e = 0; e < 8; ++e) acc[i][j][e] = 0.0f;

  stage(0, 0);
  const int nk = K >> 5;
  for (int t = 0; t < nk; ++t) {
    const int buf = t & 1;
    if (t + 1 < nk) {
      stage((t + 1) << 5, buf ^ 1);  // prefetch next tile into other buffer
      WAIT_ASYNC_4();                // wait only for current tile's 4 loads
    } else {
      WAIT_ASYNC_0();
    }
    __syncthreads();  // publish LDS tile across waves

    v16h af[4], bf[2];
#pragma unroll
    for (int ms = 0; ms < 4; ++ms)
      af[ms] = load_frag_a(&As[buf][wm * 64 + ms * 16][0], 40, lane);
#pragma unroll
    for (int ns = 0; ns < 2; ++ns)
      bf[ns] = load_frag_b(&Bs[buf][wn * 32 + ns * 16][0], 40, lane);
#pragma unroll
    for (int ms = 0; ms < 4; ++ms)
#pragma unroll
      for (int ns = 0; ns < 2; ++ns)
        acc[ms][ns] = __builtin_amdgcn_wmma_f32_16x16x32_f16(
            false, af[ms], false, bf[ns], (short)0, acc[ms][ns], false, false);

    __syncthreads();  // all reads done before this buffer is refilled
  }

  const int half = lane >> 4, nn = lane & 15;
#pragma unroll
  for (int ms = 0; ms < 4; ++ms)
#pragma unroll
    for (int ns = 0; ns < 2; ++ns)
#pragma unroll
      for (int r = 0; r < 8; ++r) {
        const int m = blockM + wm * 64 + ms * 16 + r + 8 * half;
        const int n = blockN + wn * 32 + ns * 16 + nn;
        const float v = acc[ms][ns][r];
        if constexpr (OUT_F16)
          reinterpret_cast<_Float16*>(Cout)[(size_t)m * N + n] = (_Float16)v;
        else
          reinterpret_cast<float*>(Cout)[(size_t)m * N + n] = v;
      }
}

// ---------------------------------------------------------------------------
// RoPE + head split. thread = (b,h,s,i), i = pair index 0..31.
// Qm/Km: [B][H][S][64] f16 (rotated) ; Vt: [B][H][64][S] f16 (transposed)
// ---------------------------------------------------------------------------
__global__ __launch_bounds__(256) void rope_pack(const _Float16* __restrict__ Qh,
                                                 const _Float16* __restrict__ Kh,
                                                 const _Float16* __restrict__ Vh,
                                                 _Float16* __restrict__ Qm,
                                                 _Float16* __restrict__ Km,
                                                 _Float16* __restrict__ Vt) {
  const int idx = blockIdx.x * 256 + threadIdx.x;
  const int i = idx & 31;
  const int s = (idx >> 5) & (S_ - 1);
  const int h = (idx >> 16) & (H_ - 1);
  const int b = idx >> 20;

  const float inv = __expf(-0.14391156f * (float)(2 * i));  // theta^(-2i/64)
  const float ang = (float)s * inv;
  const float cs = __cosf(ang), sn = __sinf(ang);

  const size_t src = ((size_t)(b * S_ + s) << 10) + h * DH_ + 2 * i;
  const size_t dst = (((size_t)(b * H_ + h) * S_ + s) << 6) + 2 * i;
  {
    const float x1 = (float)Qh[src], x2 = (float)Qh[src + 1];
    Qm[dst]     = (_Float16)(x1 * cs - x2 * sn);
    Qm[dst + 1] = (_Float16)(x2 * cs + x1 * sn);
  }
  {
    const float x1 = (float)Kh[src], x2 = (float)Kh[src + 1];
    Km[dst]     = (_Float16)(x1 * cs - x2 * sn);
    Km[dst + 1] = (_Float16)(x2 * cs + x1 * sn);
  }
  const size_t vdst = ((size_t)((b * H_ + h) * DH_ + 2 * i)) * S_ + s;
  Vt[vdst]      = Vh[src];
  Vt[vdst + S_] = Vh[src + 1];
}

// ---------------------------------------------------------------------------
// Flash attention (causal, streaming softmax). grid=(S/128, B*H), 256 thr.
// Wave w owns query tile [qbase, qbase+16); 8 waves share 32-key K/V LDS tiles
// staged with global_load_async_to_lds_b128 (ASYNCcnt + barrier publish).
// Interior (below-diagonal) tiles skip all causal-mask VALU work; only the
// 1-2 diagonal tiles per wave run the masked path (branch is wave-uniform).
// ---------------------------------------------------------------------------
__global__ __launch_bounds__(256) void attn_fwd(const _Float16* __restrict__ Q,
                                                const _Float16* __restrict__ Km,
                                                const _Float16* __restrict__ Vt,
                                                _Float16* __restrict__ O) {
  __shared__ _Float16 Ks[32][72];     // 32 keys x 64 dh ; 144B rows (16B-aligned)
  __shared__ _Float16 Vs[64][40];     // 64 dh x 32 keys ;  80B rows (16B-aligned)
  __shared__ _Float16 Ps[8][16][36];  // per-wave P transpose scratch

  const int tid  = threadIdx.x;
  const int lane = tid & 31;
  const int wave = tid >> 5;
  const int bh = blockIdx.y;
  const int b = bh >> 4, h = bh & 15;
  const int qblock = blockIdx.x;
  const int qbase = qblock * 128 + wave * 16;

  const _Float16* Qbh = Q  + (size_t)bh * S_ * DH_;
  const _Float16* Kbh = Km + (size_t)bh * S_ * DH_;
  const _Float16* Vbh = Vt + (size_t)bh * DH_ * S_;

  const uint32_t ldsK = (uint32_t)(uintptr_t)&Ks[0][0];
  const uint32_t ldsV = (uint32_t)(uintptr_t)&Vs[0][0];

  // Q tile 16x64 -> two A fragments; fold in 1/sqrt(64) scale.
  v16h qf[2];
#pragma unroll
  for (int c = 0; c < 2; ++c) {
    qf[c] = load_frag_a(Qbh + (size_t)qbase * DH_ + c * 32, DH_, lane);
#pragma unroll
    for (int e = 0; e < 16; ++e) qf[c][e] = qf[c][e] * (_Float16)0.125f;
  }

  v8f o[4];
#pragma unroll
  for (int c = 0; c < 4; ++c)
#pragma unroll
    for (int e = 0; e < 8; ++e) o[c][e] = 0.0f;
  float mrow[8], lsum[8];
#pragma unroll
  for (int r = 0; r < 8; ++r) { mrow[r] = -3.0e38f; lsum[r] = 0.0f; }

  const int half = lane >> 4, nn = lane & 15;
  const int ntiles = (qblock + 1) * 4;  // uniform across workgroup

  for (int t = 0; t < ntiles; ++t) {
    const int kbase = t * 32;
    __syncthreads();  // previous tile fully consumed before refill
    {
      // V tile: 64 rows x 4 chunks = 256 x 16B -> one async per thread
      const int vr = tid >> 2, vc = tid & 3;
      async_ld_b128(ldsV + (uint32_t)(vr * 80 + vc * 16),
                    (const char*)Vbh + ((size_t)vr * S_ + kbase + vc * 8) * 2);
      // K tile: 32 rows x 4 chunks = 128 x 16B -> waves 0..3 (uniform branch)
      if (tid < 128) {
        const int kr = tid >> 2, kc = tid & 3;
        async_ld_b128(ldsK + (uint32_t)(kr * 144 + kc * 16),
                      (const char*)Kbh + ((size_t)(kbase + kr) * DH_ + kc * 8) * 2);
      }
      WAIT_ASYNC_0();
    }
    __syncthreads();
    if (kbase > qbase + 15) continue;        // wave-uniform; no barrier skipped
    const bool masked = (kbase + 31) > qbase;  // diagonal tile? (wave-uniform)

    // S = Q * K^T : 16x32 scores (2 C-frags), accumulate over dh halves
    v8f sc[2];
#pragma unroll
    for (int j = 0; j < 2; ++j) {
#pragma unroll
      for (int e = 0; e < 8; ++e) sc[j][e] = 0.0f;
#pragma unroll
      for (int c = 0; c < 2; ++c) {
        const v16h bf = load_frag_b(&Ks[j * 16][c * 32], 72, lane);
        sc[j] = __builtin_amdgcn_wmma_f32_16x16x32_f16(false, qf[c], false, bf, (short)0,
                                                       sc[j], false, false);
      }
    }

    // causal mask only on diagonal tiles
    if (masked) {
#pragma unroll
      for (int r = 0; r < 8; ++r) {
        const int q = qbase + r + 8 * half;
        if (kbase + nn > q) sc[0][r] = -3.0e38f;
        if (kbase + 16 + nn > q) sc[1][r] = -3.0e38f;
      }
    }

    // per-row streaming softmax (rows live in 16-lane halves)
    float tmax[8];
#pragma unroll
    for (int r = 0; r < 8; ++r) tmax[r] = fmaxf(sc[0][r], sc[1][r]);
#pragma unroll
    for (int msk = 1; msk < 16; msk <<= 1)
#pragma unroll
      for (int r = 0; r < 8; ++r) tmax[r] = fmaxf(tmax[r], __shfl_xor(tmax[r], msk, 32));

    float alpha[8], tsum[8];
#pragma unroll
    for (int r = 0; r < 8; ++r) {
      const float mnew = fmaxf(mrow[r], tmax[r]);
      alpha[r] = __expf(mrow[r] - mnew);
      mrow[r] = mnew;
      const float p0 = __expf(sc[0][r] - mnew);
      const float p1 = __expf(sc[1][r] - mnew);
      sc[0][r] = p0;
      sc[1][r] = p1;
      tsum[r] = p0 + p1;
    }
#pragma unroll
    for (int msk = 1; msk < 16; msk <<= 1)
#pragma unroll
      for (int r = 0; r < 8; ++r) tsum[r] += __shfl_xor(tsum[r], msk, 32);
#pragma unroll
    for (int r = 0; r < 8; ++r) lsum[r] = lsum[r] * alpha[r] + tsum[r];
#pragma unroll
    for (int c = 0; c < 4; ++c)
#pragma unroll
      for (int r = 0; r < 8; ++r) o[c][r] *= alpha[r];

    // C-layout -> A-layout transpose of P through per-wave LDS scratch
#pragma unroll
    for (int j = 0; j < 2; ++j)
#pragma unroll
      for (int r = 0; r < 8; ++r)
        Ps[wave][r + 8 * half][j * 16 + nn] = (_Float16)sc[j][r];

    const v16h pf = load_frag_a(&Ps[wave][0][0], 36, lane);
    // O += P(16x32) * V(32x64) : 4 output N-chunks
#pragma unroll
    for (int c = 0; c < 4; ++c) {
      const v16h vf = load_frag_b(&Vs[c * 16][0], 40, lane);
      o[c] = __builtin_amdgcn_wmma_f32_16x16x32_f16(false, pf, false, vf, (short)0, o[c],
                                                    false, false);
    }
  }

  // normalize and write merged-head f16 output [B*S][D]
#pragma unroll
  for (int c = 0; c < 4; ++c)
#pragma unroll
    for (int r = 0; r < 8; ++r) {
      const float val = o[c][r] / lsum[r];
      const int m = qbase + r + 8 * half;
      O[((size_t)(b * S_ + m) << 10) + h * DH_ + c * 16 + nn] = (_Float16)val;
    }
}

// ---------------------------------------------------------------------------
// Launcher
// ---------------------------------------------------------------------------
extern "C" void kernel_launch(void* const* d_in, const int* in_sizes, int n_in,
                              void* d_out, int out_size, void* d_ws, size_t ws_size,
                              hipStream_t stream) {
  (void)in_sizes; (void)n_in; (void)out_size; (void)ws_size;
  const float* x  = (const float*)d_in[0];
  const float* Wq = (const float*)d_in[1];
  const float* Wk = (const float*)d_in[2];
  const float* Wv = (const float*)d_in[3];
  const float* Wo = (const float*)d_in[4];
  float* out = (float*)d_out;

  char* p = (char*)d_ws;
  auto carve = [&](size_t bytes) {
    void* r = (void*)p;
    p += (bytes + 255) & ~(size_t)255;
    return r;
  };
  const size_t actB = (size_t)BS_ * D_ * sizeof(_Float16);  // 16 MB
  const size_t wB   = (size_t)D_ * D_ * sizeof(_Float16);   //  2 MB
  _Float16* xh   = (_Float16*)carve(actB);
  _Float16* WqT  = (_Float16*)carve(wB);
  _Float16* WkT  = (_Float16*)carve(wB);
  _Float16* WvT  = (_Float16*)carve(wB);
  _Float16* WoT  = (_Float16*)carve(wB);
  _Float16* Qh   = (_Float16*)carve(actB);
  _Float16* Kh   = (_Float16*)carve(actB);
  _Float16* Vh   = (_Float16*)carve(actB);
  _Float16* Qm   = (_Float16*)carve(actB);
  _Float16* Km   = (_Float16*)carve(actB);
  _Float16* Vt   = (_Float16*)carve(actB);
  _Float16* attn = (_Float16*)carve(actB);

  const int nAct = BS_ * D_;
  cvt_f32_f16<<<nAct / 256, 256, 0, stream>>>(x, xh, nAct);
  transpose_cvt<<<(D_ * D_) / 256, 256, 0, stream>>>(Wq, WqT);
  transpose_cvt<<<(D_ * D_) / 256, 256, 0, stream>>>(Wk, WkT);
  transpose_cvt<<<(D_ * D_) / 256, 256, 0, stream>>>(Wv, WvT);
  transpose_cvt<<<(D_ * D_) / 256, 256, 0, stream>>>(Wo, WoT);

  dim3 ggrid(D_ / 128, BS_ / 128);
  gemm_wmma<true><<<ggrid, 256, 0, stream>>>(xh, WqT, Qh, BS_, D_, D_);
  gemm_wmma<true><<<ggrid, 256, 0, stream>>>(xh, WkT, Kh, BS_, D_, D_);
  gemm_wmma<true><<<ggrid, 256, 0, stream>>>(xh, WvT, Vh, BS_, D_, D_);

  rope_pack<<<(B_ * H_ * S_ * 32) / 256, 256, 0, stream>>>(Qh, Kh, Vh, Qm, Km, Vt);

  attn_fwd<<<dim3(S_ / 128, B_ * H_), 256, 0, stream>>>(Qm, Km, Vt, attn);

  gemm_wmma<false><<<ggrid, 256, 0, stream>>>(attn, WoT, out, BS_, D_, D_);
}
